// Attention_62440234549585
// MI455X (gfx1250) — compile-verified, matrix operand order checked by reading, not run on hardware
//
#include <hip/hip_runtime.h>
#include <stdint.h>

#define BATCH 8
#define SEQ   1024
#define CDIM  768
#define NH    12
#define HD    64
#define MTOT  (BATCH * SEQ)   // 8192
#define OQKV  (3 * CDIM)      // 2304
// 64^-0.5 * log2(e): folded into Q so softmax exponentials are bare v_exp_f32.
#define QSCALE_L2E (0.125f * 1.4426950408889634f)

typedef __attribute__((ext_vector_type(16))) __bf16 v16bf;
typedef __attribute__((ext_vector_type(8)))  __bf16 v8bf;
typedef __attribute__((ext_vector_type(8)))  float  v8f;
typedef __attribute__((ext_vector_type(4)))  float  v4f;

#define CAT16(lo, hi) __builtin_shufflevector((lo), (hi), 0,1,2,3,4,5,6,7,8,9,10,11,12,13,14,15)

__device__ __forceinline__ __bf16 f2bf(float f) {
  union { float f; uint32_t u; } x; x.f = f;
  uint32_t r = x.u + 0x7FFFu + ((x.u >> 16) & 1u);   // round-to-nearest-even
  uint16_t h = (uint16_t)(r >> 16);
  __bf16 b;
  __builtin_memcpy(&b, &h, sizeof(h));
  return b;
}

// ---------------------------------------------------------------------------
// fp32 -> bf16 conversion, 8 elements / thread (b128 in, b128 out)
// ---------------------------------------------------------------------------
__global__ void cvt_kernel(const float* __restrict__ in, __bf16* __restrict__ out, int n8) {
  int i = blockIdx.x * blockDim.x + threadIdx.x;
  if (i < n8) {
    v4f a = *(const v4f*)(in + (size_t)i * 8);
    v4f b = *(const v4f*)(in + (size_t)i * 8 + 4);
    v8bf o;
#pragma unroll
    for (int j = 0; j < 4; ++j) { o[j] = f2bf(a[j]); o[j + 4] = f2bf(b[j]); }
    *(v8bf*)(out + (size_t)i * 8) = o;
  }
}

// ---------------------------------------------------------------------------
// Tiled bf16 WMMA GEMM: C[m,n] = sum_k A[m,k] * W[n,k]
// Block tile 128x128x32, 8 waves (4 in M x 2 in N), wave tile 32x64.
// Register double-buffering of the global->LDS staging.
// MODE 0: QKV epilogue (scatter to Q scaled / K / V-transposed, bf16)
// MODE 1: proj epilogue (fp32 out + bias)
// ---------------------------------------------------------------------------
#define TM 128
#define TN 128
#define TK 32
#define LK 40   // padded LDS row stride (bf16): 80B, conflict-free b128 reads

template<int MODE>
__global__ __launch_bounds__(256)
void gemm_kernel(const __bf16* __restrict__ A, const __bf16* __restrict__ W, int K,
                 __bf16* __restrict__ qout, __bf16* __restrict__ kout,
                 __bf16* __restrict__ vout,
                 float* __restrict__ pout, const float* __restrict__ bias) {
  __shared__ __bf16 As[TM][LK];
  __shared__ __bf16 Bs[TN][LK];

  const int tid  = threadIdx.x;
  const int lane = tid & 31;
  const int wave = tid >> 5;
  const int wm   = wave & 3;           // wave row 0..3
  const int wn   = wave >> 2;          // wave col 0..1
  const int r16  = lane & 15;
  const int hi8  = (lane & 16) ? 8 : 0;

  const int m0 = blockIdx.y * TM;
  const int n0 = blockIdx.x * TN;

  // Two 8-elem chunks per thread for each of A and B (512 chunks / 256 threads).
  const int rowC0 = tid >> 2,          kcC0 = (tid & 3) * 8;
  const int rowC1 = (tid + 256) >> 2,  kcC1 = ((tid + 256) & 3) * 8;

  const __bf16* Arow0 = A + (size_t)(m0 + rowC0) * K + kcC0;
  const __bf16* Arow1 = A + (size_t)(m0 + rowC1) * K + kcC1;
  const __bf16* Wrow0 = W + (size_t)(n0 + rowC0) * K + kcC0;
  const __bf16* Wrow1 = W + (size_t)(n0 + rowC1) * K + kcC1;

  v8bf ra0 = *(const v8bf*)(Arow0);
  v8bf ra1 = *(const v8bf*)(Arow1);
  v8bf rb0 = *(const v8bf*)(Wrow0);
  v8bf rb1 = *(const v8bf*)(Wrow1);

  v8f acc[2][4] = {};

  for (int kt = 0; kt < K; kt += TK) {
    // Commit staged registers to LDS.
    *(v8bf*)&As[rowC0][kcC0] = ra0;
    *(v8bf*)&As[rowC1][kcC1] = ra1;
    *(v8bf*)&Bs[rowC0][kcC0] = rb0;
    *(v8bf*)&Bs[rowC1][kcC1] = rb1;
    __syncthreads();

    // Prefetch next k-tile while this tile is consumed.
    if (kt + TK < K) {
      ra0 = *(const v8bf*)(Arow0 + kt + TK);
      ra1 = *(const v8bf*)(Arow1 + kt + TK);
      rb0 = *(const v8bf*)(Wrow0 + kt + TK);
      rb1 = *(const v8bf*)(Wrow1 + kt + TK);
    }

    v16bf af[2], bf[4];
#pragma unroll
    for (int sm = 0; sm < 2; ++sm) {
      int row = wm * 32 + sm * 16 + r16;
      v8bf lo = *(const v8bf*)&As[row][hi8];
      v8bf hi = *(const v8bf*)&As[row][16 + hi8];
      af[sm] = CAT16(lo, hi);
    }
#pragma unroll
    for (int sn = 0; sn < 4; ++sn) {
      int col = wn * 64 + sn * 16 + r16;
      v8bf lo = *(const v8bf*)&Bs[col][hi8];
      v8bf hi = *(const v8bf*)&Bs[col][16 + hi8];
      bf[sn] = CAT16(lo, hi);
    }
#pragma unroll
    for (int sm = 0; sm < 2; ++sm)
#pragma unroll
      for (int sn = 0; sn < 4; ++sn)
        acc[sm][sn] = __builtin_amdgcn_wmma_f32_16x16x32_bf16(
            false, af[sm], false, bf[sn], (short)0, acc[sm][sn], false, false);
    __syncthreads();
  }

  // Epilogue
  const int mBase = m0 + wm * 32;
  const int nBase = n0 + wn * 64;
#pragma unroll
  for (int sm = 0; sm < 2; ++sm) {
#pragma unroll
    for (int sn = 0; sn < 4; ++sn) {
#pragma unroll
      for (int r = 0; r < 8; ++r) {
        int m = mBase + sm * 16 + r + hi8;
        int n = nBase + sn * 16 + r16;
        float val = acc[sm][sn][r];
        if (MODE == 0) {
          int t   = n / CDIM;
          int rem = n - t * CDIM;
          int h   = rem >> 6;
          int d   = rem & 63;
          int b   = m >> 10;
          int s   = m & (SEQ - 1);
          if (t == 0) {
            qout[(((size_t)b * NH + h) * SEQ + s) * HD + d] = f2bf(val * QSCALE_L2E);
          } else if (t == 1) {
            kout[(((size_t)b * NH + h) * SEQ + s) * HD + d] = f2bf(val);
          } else {  // V stored transposed: [BH, 64, SEQ]
            vout[(((size_t)b * NH + h) * HD + d) * SEQ + s] = f2bf(val);
          }
        } else {
          pout[(size_t)m * CDIM + n] = val + bias[n];
        }
      }
    }
  }
}

// ---------------------------------------------------------------------------
// Flash attention: one wave per 16-query tile, 32 keys per step.
// Softmax stats computed in A-layout lane space (2 cross-lane ops per step),
// P packed directly into the WMMA A-fragment. Exponentials are base-2
// (log2e folded into Q), so each is a single v_exp_f32.
// Q:[BH,SEQ,64] (pre-scaled), K:[BH,SEQ,64], Vt:[BH,64,SEQ], out:[B,SEQ,768] bf16
// ---------------------------------------------------------------------------
#define SLK 36  // padded fp32 LDS row stride (144B): conflict-free

__global__ __launch_bounds__(128)
void attn_kernel(const __bf16* __restrict__ Q, const __bf16* __restrict__ Km,
                 const __bf16* __restrict__ Vt, __bf16* __restrict__ Ao) {
  __shared__ float Sl[4][16][SLK];   // per-wave fp32 S staging (C- -> A-layout)

  const int lane = threadIdx.x & 31;
  const int wave = threadIdx.x >> 5;
  const int r16  = lane & 15;
  const int hi8  = (lane & 16) ? 8 : 0;

  const int bh = blockIdx.y;                 // 0..95
  const int q0 = blockIdx.x * 64 + wave * 16;

  const __bf16* qp = Q  + (size_t)bh * SEQ * HD;
  const __bf16* kp = Km + (size_t)bh * SEQ * HD;
  const __bf16* vp = Vt + (size_t)bh * HD * SEQ;

  // Q A-fragments (d=0..31, d=32..63) pinned in registers.
  v16bf qf[2];
  {
    const __bf16* qrow = qp + (size_t)(q0 + r16) * HD;
#pragma unroll
    for (int f = 0; f < 2; ++f) {
      v8bf lo = *(const v8bf*)(qrow + f * 32 + hi8);
      v8bf hi = *(const v8bf*)(qrow + f * 32 + 16 + hi8);
      qf[f] = CAT16(lo, hi);
    }
  }

  // Per-lane online-softmax state for row r16 (replicated in lane^16 partner).
  float mloc = -1e30f, lloc = 0.0f;
  v8f oacc[4] = {};

  for (int j0 = 0; j0 < SEQ; j0 += 32) {
    // ---- S = Q K^T : two 16x16 tiles over the 32-key slab
    v8f s[2];
#pragma unroll
    for (int sn = 0; sn < 2; ++sn) {
      const __bf16* krow = kp + (size_t)(j0 + sn * 16 + r16) * HD;
      v16bf kf0 = CAT16(*(const v8bf*)(krow + hi8),      *(const v8bf*)(krow + 16 + hi8));
      v16bf kf1 = CAT16(*(const v8bf*)(krow + 32 + hi8), *(const v8bf*)(krow + 48 + hi8));
      v8f c = {};
      c = __builtin_amdgcn_wmma_f32_16x16x32_bf16(false, qf[0], false, kf0,
                                                  (short)0, c, false, false);
      c = __builtin_amdgcn_wmma_f32_16x16x32_bf16(false, qf[1], false, kf1,
                                                  (short)0, c, false, false);
      s[sn] = c;
    }

    // ---- Transpose S through LDS: C-layout -> A-layout (fp32)
#pragma unroll
    for (int r = 0; r < 8; ++r) {
      Sl[wave][r + hi8][r16]      = s[0][r];
      Sl[wave][r + hi8][16 + r16] = s[1][r];
    }
    __builtin_amdgcn_wave_barrier();

    // Lane holds row r16, k = hi8..hi8+7 and 16+hi8..16+hi8+7 (A-frag order).
    float vals[16];
    {
      const float* srow = &Sl[wave][r16][0];
      v4f c0 = *(const v4f*)(srow + hi8);
      v4f c1 = *(const v4f*)(srow + hi8 + 4);
      v4f c2 = *(const v4f*)(srow + 16 + hi8);
      v4f c3 = *(const v4f*)(srow + 16 + hi8 + 4);
#pragma unroll
      for (int i = 0; i < 4; ++i) {
        vals[i] = c0[i]; vals[i + 4] = c1[i];
        vals[i + 8] = c2[i]; vals[i + 12] = c3[i];
      }
    }
    __builtin_amdgcn_wave_barrier();

    // ---- In-lane softmax stats; single xor-16 exchange with partner lane.
    float mx = vals[0];
#pragma unroll
    for (int i = 1; i < 16; ++i) mx = fmaxf(mx, vals[i]);
    mx = fmaxf(mx, __shfl_xor(mx, 16, 32));
    float mnew = fmaxf(mloc, mx);
    float corr = __builtin_amdgcn_exp2f(mloc - mnew);   // base-2: bare v_exp_f32
    mloc = mnew;

    float sum = 0.0f;
    v16bf pf;
#pragma unroll
    for (int i = 0; i < 16; ++i) {
      float p = __builtin_amdgcn_exp2f(vals[i] - mnew); // bare v_exp_f32
      sum += p;
      pf[i] = f2bf(p);
    }
    sum += __shfl_xor(sum, 16, 32);
    lloc = lloc * corr + sum;

    // ---- Broadcast per-row rescale factor to C-layout and rescale O.
    float corrRow[8];
#pragma unroll
    for (int r = 0; r < 8; ++r) corrRow[r] = __shfl(corr, r + hi8, 32);
#pragma unroll
    for (int sn = 0; sn < 4; ++sn)
#pragma unroll
      for (int r = 0; r < 8; ++r) oacc[sn][r] *= corrRow[r];

    // ---- O += P * V  (V transposed: contiguous along keys)
#pragma unroll
    for (int sn = 0; sn < 4; ++sn) {
      const __bf16* vrow = vp + (size_t)(sn * 16 + r16) * SEQ + j0;
      v16bf vf = CAT16(*(const v8bf*)(vrow + hi8), *(const v8bf*)(vrow + 16 + hi8));
      oacc[sn] = __builtin_amdgcn_wmma_f32_16x16x32_bf16(false, pf, false, vf,
                                                         (short)0, oacc[sn], false, false);
    }
  }

  // ---- Normalize (rcp+mul) and write [B, SEQ, 768] bf16
  const int b = bh / NH;
  const int h = bh % NH;
  float lInv[8];
#pragma unroll
  for (int r = 0; r < 8; ++r)
    lInv[r] = __builtin_amdgcn_rcpf(__shfl(lloc, r + hi8, 32));
#pragma unroll
  for (int sn = 0; sn < 4; ++sn) {
#pragma unroll
    for (int r = 0; r < 8; ++r) {
      int m = q0 + r + hi8;
      int d = sn * 16 + r16;
      float o = oacc[sn][r] * lInv[r];
      Ao[((size_t)b * SEQ + m) * CDIM + h * HD + d] = f2bf(o);
    }
  }
}

// ---------------------------------------------------------------------------
extern "C" void kernel_launch(void* const* d_in, const int* in_sizes, int n_in,
                              void* d_out, int out_size, void* d_ws, size_t ws_size,
                              hipStream_t stream) {
  const float* x      = (const float*)d_in[0];
  const float* qkv_w  = (const float*)d_in[1];
  const float* proj_w = (const float*)d_in[2];
  const float* proj_b = (const float*)d_in[3];
  float* out = (float*)d_out;

  uint8_t* ws = (uint8_t*)d_ws;
  size_t off = 0;
  auto alloc_bf16 = [&](size_t elems) -> __bf16* {
    __bf16* p = (__bf16*)(ws + off);
    off += ((elems * 2) + 255) & ~(size_t)255;
    return p;
  };
  __bf16* xb  = alloc_bf16((size_t)MTOT * CDIM);   // x in bf16 (reused as attn out)
  __bf16* wqb = alloc_bf16((size_t)OQKV * CDIM);   // qkv_w bf16
  __bf16* wpb = alloc_bf16((size_t)CDIM * CDIM);   // proj_w bf16
  __bf16* qb  = alloc_bf16((size_t)BATCH * NH * SEQ * HD);
  __bf16* kb  = alloc_bf16((size_t)BATCH * NH * SEQ * HD);
  __bf16* vtb = alloc_bf16((size_t)BATCH * NH * HD * SEQ);
  __bf16* aob = xb;  // x no longer needed after QKV GEMM -> reuse slot

  int n1 = (MTOT * CDIM) / 8;
  cvt_kernel<<<(n1 + 255) / 256, 256, 0, stream>>>(x, xb, n1);
  int n2 = (OQKV * CDIM) / 8;
  cvt_kernel<<<(n2 + 255) / 256, 256, 0, stream>>>(qkv_w, wqb, n2);
  int n3 = (CDIM * CDIM) / 8;
  cvt_kernel<<<(n3 + 255) / 256, 256, 0, stream>>>(proj_w, wpb, n3);

  dim3 g1(OQKV / TN, MTOT / TM);   // 18 x 64
  gemm_kernel<0><<<g1, 256, 0, stream>>>(xb, wqb, CDIM, qb, kb, vtb, nullptr, nullptr);

  dim3 g2(SEQ / 64, BATCH * NH);   // 16 x 96, 4 waves/block
  attn_kernel<<<g2, 128, 0, stream>>>(qb, kb, vtb, aob);

  dim3 g3(CDIM / TN, MTOT / TM);   // 6 x 64
  gemm_kernel<1><<<g3, 256, 0, stream>>>(aob, wpb, CDIM, nullptr, nullptr, nullptr,
                                         out, proj_b);
}
